// GATFraudDetector_4930622456179
// MI455X (gfx1250) — matrix operand order repaired and checked by reading, not compiled, not run on hardware
//
#include <hip/hip_runtime.h>
#include <hip/hip_bf16.h>
#include <math.h>

// ---------------------------------------------------------------------------
// GAT fraud detector for MI455X (gfx1250, wave32, WMMA).
// GEMMs: v_wmma_f32_16x16x32_bf16, f32 accumulate, 32-row macro-tile/wave,
// b128 A loads + v_cvt_pk_bf16_f32 packing, bank-conflict-free padded LDS B.
// Irregular scatter/softmax parts: VALU + global f32 atomics.
// Workspace requirement: ~600 MB (edge tensors dominate).
// ---------------------------------------------------------------------------

#define HID 64
#define HEADS 4
#define NEG_SLOPE 0.2f
#define LN_EPS 1e-5f
#define LDS_PAD 8            // shorts; row stride K+8 -> conflict-free banks

typedef __attribute__((ext_vector_type(16))) __bf16    bf16x16;
typedef __attribute__((ext_vector_type(8)))  float     f32x8;
typedef __attribute__((ext_vector_type(4)))  float     f32x4;
typedef __attribute__((ext_vector_type(4)))  unsigned  u32x4;

union Frag { unsigned u[8]; u32x4 q[2]; bf16x16 v; };

// pack two f32 -> packed bf16 pair (low = first arg), round-to-nearest-even
__device__ inline unsigned pack2(float a, float b) {
  unsigned r;
  asm volatile("v_cvt_pk_bf16_f32 %0, %1, %2" : "=v"(r) : "v"(a), "v"(b));
  return r;
}

// scalar f32 -> bf16 bits (RNE), used once for weight staging
__device__ inline unsigned short bf16bits(float f) {
  unsigned u = __float_as_uint(f);
  return (unsigned short)((u + 0x7FFFu + ((u >> 16) & 1u)) >> 16);
}

// load 8 contiguous floats (16B-aligned) and pack into 4 bf16-pair dwords
__device__ inline void load8(const float* __restrict__ r, unsigned* d) {
  f32x4 q0 = *(const f32x4*)r;
  f32x4 q1 = *(const f32x4*)(r + 4);
  d[0] = pack2(q0.x, q0.y); d[1] = pack2(q0.z, q0.w);
  d[2] = pack2(q1.x, q1.y); d[3] = pack2(q1.z, q1.w);
}

// order-preserving float<->uint map for atomicMax-based segment max
__device__ inline unsigned ford(float f) {
  unsigned u = __float_as_uint(f);
  return (u & 0x80000000u) ? ~u : (u | 0x80000000u);
}
__device__ inline float funord(unsigned o) {
  unsigned b = (o & 0x80000000u) ? (o ^ 0x80000000u) : ~o;
  return __uint_as_float(b);
}
#define ORD_NEG_INF 0x007FFFFFu   // ford(-inf)

// ---------------------------------------------------------------------------
// Generic WMMA GEMM:  Out[M,Nout] = act( A[M,K] @ W[Nout,K]^T + bias )
// GATHER: A row m = concat(Hn[src[m]], Hn[dst[m]], Ea[m])  (K == 192)
// ACCUM : Out += result (residual update of ea)
// 32-row macro-tile per wave (2 x 16-row WMMA subtiles sharing B fragment);
// 8 waves / block; weights staged in LDS as bf16 with padded rows.
// ---------------------------------------------------------------------------
template <bool GATHER, bool RELU, bool ACCUM>
__global__ __launch_bounds__(256) void gemm_kernel(
    const float* __restrict__ A, const float* __restrict__ Hn,
    const float* __restrict__ Ea, const int* __restrict__ src,
    const int* __restrict__ dst, const float* __restrict__ W,
    const float* __restrict__ bias, float* __restrict__ Out,
    int M, int K, int Nout)
{
  __shared__ unsigned short sW[64 * (192 + LDS_PAD)];   // 25.6 KB max
  const int stride = K + LDS_PAD;
  const int lane = threadIdx.x & 31;
  const int wid  = threadIdx.x >> 5;

  // stage weights as bf16, one wave per row, padded row stride
  for (int n = wid; n < Nout; n += 8)
    for (int k = lane; k < K; k += 32)
      sW[n * stride + k] = bf16bits(W[n * K + k]);
  __syncthreads();

  const int tile = blockIdx.x * 8 + wid;                 // 32-row macro-tile
  if (tile >= ((M + 31) >> 5)) return;                   // wave-uniform exit
  const int m0 = tile << 5;
  const int lm = lane & 15;
  const int hi = lane >> 4;

  // per-subtile source row pointers (rows clamped to keep EXEC all-ones)
  const float *rS[2], *rD[2], *rE[2];
#pragma unroll
  for (int s = 0; s < 2; ++s) {
    int m = m0 + s * 16 + lm;
    if (m >= M) m = M - 1;
    if (GATHER) {
      rS[s] = Hn + (size_t)src[m] * HID;
      rD[s] = Hn + (size_t)dst[m] * HID;
      rE[s] = Ea + (size_t)m * HID;
    } else {
      rS[s] = A + (size_t)m * K;
    }
  }

  const bool full = (m0 + 32) <= M;                      // wave-uniform

  for (int n0 = 0; n0 < Nout; n0 += 16) {
    f32x8 acc[2] = {{}, {}};
    const unsigned short* wRow = sW + (n0 + lm) * stride;
    for (int k0 = 0; k0 < K; k0 += 32) {
      // B fragment: 16 consecutive bf16 from padded LDS row (2x ds_load_b128)
      Frag fb;
      const unsigned short* wp = wRow + k0 + hi * 16;
      fb.q[0] = *(const u32x4*)wp;
      fb.q[1] = *(const u32x4*)(wp + 8);
      const int kaLo = k0 + hi * 16;
      const int kaHi = kaLo + 8;
#pragma unroll
      for (int s = 0; s < 2; ++s) {
        Frag fa;
        if (GATHER) {
          // each 8-float run stays inside one 64-wide gathered segment
          const float* pLo = (kaLo < 64) ? (rS[s] + kaLo)
                           : (kaLo < 128) ? (rD[s] + (kaLo - 64))
                                          : (rE[s] + (kaLo - 128));
          const float* pHi = (kaHi < 64) ? (rS[s] + kaHi)
                           : (kaHi < 128) ? (rD[s] + (kaHi - 64))
                                          : (rE[s] + (kaHi - 128));
          load8(pLo, &fa.u[0]);
          load8(pHi, &fa.u[4]);
        } else {
          load8(rS[s] + kaLo, &fa.u[0]);
          load8(rS[s] + kaHi, &fa.u[4]);
        }
        acc[s] = __builtin_amdgcn_wmma_f32_16x16x32_bf16(
            false, fa.v, false, fb.v, (short)0, acc[s], false, false);
      }
    }
    const float bv = bias[n0 + lm];
#pragma unroll
    for (int s = 0; s < 2; ++s) {
#pragma unroll
      for (int r = 0; r < 8; ++r) {
        int mr = m0 + s * 16 + hi * 8 + r;               // C/D: M = r + 8*hi
        float val = acc[s][r] + bv;
        if (RELU) val = fmaxf(val, 0.f);
        size_t o = (size_t)mr * Nout + n0 + lm;
        if (full) {
          if (ACCUM) Out[o] += val; else Out[o] = val;
        } else if (mr < M) {
          if (ACCUM) Out[o] += val; else Out[o] = val;
        }
      }
    }
  }
}

// ---------------------------------------------------------------------------
// Fill helpers
// ---------------------------------------------------------------------------
__global__ void fill_f32(float* __restrict__ p, float v, long n) {
  long g = (long)blockIdx.x * blockDim.x + threadIdx.x;
  if (g < n) p[g] = v;
}
__global__ void fill_u32(unsigned* __restrict__ p, unsigned v, long n) {
  long g = (long)blockIdx.x * blockDim.x + threadIdx.x;
  if (g < n) p[g] = v;
}

// deg + scatter-sum of edge_attr into loop_attr (per destination node)
__global__ void scatter_loop_attr(const float* __restrict__ ea,
                                  const int* __restrict__ dst,
                                  float* __restrict__ loopA,
                                  float* __restrict__ deg, long n) {
  long g = (long)blockIdx.x * blockDim.x + threadIdx.x;
  if (g >= n) return;
  long e = g >> 6; int f = (int)(g & 63);
  int d = dst[e];
  atomicAdd(&loopA[(size_t)d * HID + f], ea[g]);
  if (f == 0) atomicAdd(&deg[d], 1.0f);
}

__global__ void finalize_loop_attr(float* __restrict__ loopA,
                                   const float* __restrict__ deg, long n) {
  long g = (long)blockIdx.x * blockDim.x + threadIdx.x;
  if (g >= n) return;
  loopA[g] /= fmaxf(deg[g >> 6], 1.0f);
}

// Wa[k,h] = sum_c edge_W[h*16+c, k] * a_e[h,c]  (fold attention into weight)
__global__ void wa_kernel(const float* __restrict__ edgW,
                          const float* __restrict__ ae,
                          float* __restrict__ Wa) {
  int g = blockIdx.x * blockDim.x + threadIdx.x;
  if (g >= HID * HEADS) return;
  int k = g >> 2, h = g & 3;
  float s = 0.f;
  for (int c = 0; c < 16; ++c) s += edgW[(h * 16 + c) * HID + k] * ae[h * 16 + c];
  Wa[k * HEADS + h] = s;
}

// alphaE[row,h] = feat[row,:] @ Wa[:,h]   (wave per row, coalesced)
__global__ void att_edge_kernel(const float* __restrict__ feat, int count,
                                const float* __restrict__ Wa,
                                float* __restrict__ out) {
  int row = blockIdx.x * 8 + (threadIdx.x >> 5);
  int lane = threadIdx.x & 31;
  if (row >= count) return;
  const float* f = feat + (size_t)row * HID;
  float v0 = f[lane], v1 = f[lane + 32];
  float p[HEADS];
#pragma unroll
  for (int h = 0; h < HEADS; ++h)
    p[h] = v0 * Wa[lane * HEADS + h] + v1 * Wa[(lane + 32) * HEADS + h];
  for (int o = 16; o; o >>= 1)
#pragma unroll
    for (int h = 0; h < HEADS; ++h) p[h] += __shfl_down(p[h], o, 32);
  if (lane == 0)
#pragma unroll
    for (int h = 0; h < HEADS; ++h) out[(size_t)row * HEADS + h] = p[h];
}

// a_src[n,h], a_dst[n,h] from projected hh
__global__ void asd_kernel(const float* __restrict__ hh,
                           const float* __restrict__ as,
                           const float* __restrict__ ad,
                           float* __restrict__ asrc, float* __restrict__ adst,
                           int N) {
  int g = blockIdx.x * blockDim.x + threadIdx.x;
  if (g >= N * HEADS) return;
  int n = g >> 2, h = g & 3;
  const float* row = hh + (size_t)n * HID + h * 16;
  float s = 0.f, d = 0.f;
  for (int c = 0; c < 16; ++c) {
    float v = row[c];
    s += v * as[h * 16 + c];
    d += v * ad[h * 16 + c];
  }
  asrc[g] = s; adst[g] = d;
}

// pass 1: alpha = leaky_relu(a_src[s]+a_dst[d]+alphaE), segment max over d
__global__ void alpha1_kernel(const int* __restrict__ src,
                              const int* __restrict__ dst,
                              const float* __restrict__ asrc,
                              const float* __restrict__ adst,
                              const float* __restrict__ alphaE,
                              float* __restrict__ alpha,
                              unsigned* __restrict__ amaxU, int E, int N) {
  long g = (long)blockIdx.x * blockDim.x + threadIdx.x;
  if (g >= (long)(E + N) * HEADS) return;
  long idx = g >> 2; int h = (int)(g & 3);
  int s = (idx < E) ? src[idx] : (int)(idx - E);
  int d = (idx < E) ? dst[idx] : (int)(idx - E);
  float a = asrc[(size_t)s * HEADS + h] + adst[(size_t)d * HEADS + h] + alphaE[g];
  a = (a > 0.f) ? a : NEG_SLOPE * a;
  alpha[g] = a;
  atomicMax(&amaxU[(size_t)d * HEADS + h], ford(a));
}

// pass 2: ex = exp(alpha - amax[d]); den[d] += ex  (ex overwrites alpha)
__global__ void alpha2_kernel(const int* __restrict__ dst,
                              float* __restrict__ alpha,
                              const unsigned* __restrict__ amaxU,
                              float* __restrict__ den, int E, int N) {
  long g = (long)blockIdx.x * blockDim.x + threadIdx.x;
  if (g >= (long)(E + N) * HEADS) return;
  long idx = g >> 2; int h = (int)(g & 3);
  int d = (idx < E) ? dst[idx] : (int)(idx - E);
  float ex = expf(alpha[g] - funord(amaxU[(size_t)d * HEADS + h]));
  alpha[g] = ex;
  atomicAdd(&den[(size_t)d * HEADS + h], ex);
}

// pass 3: gout[d,f] += hh[s,f] * (ex / den[d,h])
__global__ void message_kernel(const int* __restrict__ src,
                               const int* __restrict__ dst,
                               const float* __restrict__ hh,
                               const float* __restrict__ alpha,
                               const float* __restrict__ den,
                               float* __restrict__ gout, int E, int N) {
  long g = (long)blockIdx.x * blockDim.x + threadIdx.x;
  if (g >= (long)(E + N) * HID) return;
  long idx = g >> 6; int f = (int)(g & 63);
  int s = (idx < E) ? src[idx] : (int)(idx - E);
  int d = (idx < E) ? dst[idx] : (int)(idx - E);
  int h = f >> 4;
  float w = alpha[(size_t)idx * HEADS + h] / den[(size_t)d * HEADS + h];
  atomicAdd(&gout[(size_t)d * HID + f], hh[(size_t)s * HID + f] * w);
}

// h += relu(layernorm(gout + gat_bias))   (wave per node)
__global__ void node_update_kernel(float* __restrict__ hbuf,
                                   const float* __restrict__ gout,
                                   const float* __restrict__ gbias,
                                   const float* __restrict__ lng,
                                   const float* __restrict__ lnb, int N) {
  int n = blockIdx.x * 8 + (threadIdx.x >> 5);
  int lane = threadIdx.x & 31;
  if (n >= N) return;
  float g0 = gout[(size_t)n * HID + lane] + gbias[lane];
  float g1 = gout[(size_t)n * HID + lane + 32] + gbias[lane + 32];
  float s = g0 + g1;
  for (int o = 16; o; o >>= 1) s += __shfl_down(s, o, 32);
  float mean = __shfl(s, 0, 32) * (1.f / HID);
  float d0 = g0 - mean, d1 = g1 - mean;
  float v = d0 * d0 + d1 * d1;
  for (int o = 16; o; o >>= 1) v += __shfl_down(v, o, 32);
  float rs = rsqrtf(__shfl(v, 0, 32) * (1.f / HID) + LN_EPS);
  float y0 = d0 * rs * lng[lane] + lnb[lane];
  float y1 = d1 * rs * lng[lane + 32] + lnb[lane + 32];
  hbuf[(size_t)n * HID + lane]      += fmaxf(y0, 0.f);
  hbuf[(size_t)n * HID + lane + 32] += fmaxf(y1, 0.f);
}

// out[e] = t2[e,:] . W3 + b3   (wave per edge)
__global__ void final_dot_kernel(const float* __restrict__ t2,
                                 const float* __restrict__ W3,
                                 const float* __restrict__ b3,
                                 float* __restrict__ out, int E) {
  int e = blockIdx.x * 8 + (threadIdx.x >> 5);
  int lane = threadIdx.x & 31;
  if (e >= E) return;
  float p = t2[(size_t)e * 32 + lane] * W3[lane];
  for (int o = 16; o; o >>= 1) p += __shfl_down(p, o, 32);
  if (lane == 0) out[e] = p + b3[0];
}

// ---------------------------------------------------------------------------
extern "C" void kernel_launch(void* const* d_in, const int* in_sizes, int n_in,
                              void* d_out, int out_size, void* d_ws,
                              size_t ws_size, hipStream_t stream) {
  (void)n_in; (void)out_size; (void)ws_size;
  const float* x         = (const float*)d_in[0];
  const float* edge_attr = (const float*)d_in[1];
  const int*   ei        = (const int*)d_in[2];
  const float* nW  = (const float*)d_in[3];  const float* nb  = (const float*)d_in[4];
  const float* eW  = (const float*)d_in[5];  const float* ebi = (const float*)d_in[6];
  const float* gat_lin_W  = (const float*)d_in[7];
  const float* gat_edge_W = (const float*)d_in[8];
  const float* att_src  = (const float*)d_in[9];
  const float* att_dst  = (const float*)d_in[10];
  const float* att_edge = (const float*)d_in[11];
  const float* gat_bias = (const float*)d_in[12];
  const float* mlp_W1 = (const float*)d_in[13]; const float* mlp_b1 = (const float*)d_in[14];
  const float* mlp_W2 = (const float*)d_in[15]; const float* mlp_b2 = (const float*)d_in[16];
  const float* ln_g = (const float*)d_in[17];   const float* ln_b = (const float*)d_in[18];
  const float* cW1 = (const float*)d_in[19];    const float* cb1 = (const float*)d_in[20];
  const float* cW2 = (const float*)d_in[21];    const float* cb2 = (const float*)d_in[22];
  const float* cW3 = (const float*)d_in[23];    const float* cb3 = (const float*)d_in[24];

  const int N = in_sizes[0] / 128;
  const int E = in_sizes[1] / 64;
  const int* src = ei;
  const int* dst = ei + E;
  float* out = (float*)d_out;

  // ---- workspace carve (~600 MB total) ----
  char* wp = (char*)d_ws;
  auto carve = [&](size_t bytes) -> void* {
    void* p = (void*)wp;
    wp += (bytes + 255) & ~(size_t)255;
    return p;
  };
  float*    h      = (float*)carve((size_t)N * HID * 4);
  float*    hh     = (float*)carve((size_t)N * HID * 4);
  float*    ea     = (float*)carve((size_t)E * HID * 4);
  float*    loopA  = (float*)carve((size_t)N * HID * 4);
  float*    deg    = (float*)carve((size_t)N * 4);
  float*    Wa     = (float*)carve((size_t)HID * HEADS * 4);
  float*    alphaE = (float*)carve((size_t)(E + N) * HEADS * 4);
  float*    asrc   = (float*)carve((size_t)N * HEADS * 4);
  float*    adst   = (float*)carve((size_t)N * HEADS * 4);
  float*    alpha  = (float*)carve((size_t)(E + N) * HEADS * 4);
  unsigned* amaxU  = (unsigned*)carve((size_t)N * HEADS * 4);
  float*    den    = (float*)carve((size_t)N * HEADS * 4);
  float*    gout   = (float*)carve((size_t)N * HID * 4);
  float*    t      = (float*)carve((size_t)E * HID * 4);
  float*    t2     = (float*)carve((size_t)E * 32 * 4);
  float*    zbias  = (float*)carve(HID * 4);

  auto gb = [](long n) { return dim3((unsigned)((n + 255) / 256)); };
  const dim3 ngrid((((N + 31) / 32) + 7) / 8);   // 32-row macro-tiles, 8 waves/blk
  const dim3 egrid((((E + 31) / 32) + 7) / 8);

  fill_f32<<<1, 64, 0, stream>>>(zbias, 0.f, HID);

  // input embeddings
  gemm_kernel<false, false, false><<<ngrid, 256, 0, stream>>>(
      x, nullptr, nullptr, nullptr, nullptr, nW, nb, h, N, 128, HID);
  gemm_kernel<false, false, false><<<egrid, 256, 0, stream>>>(
      edge_attr, nullptr, nullptr, nullptr, nullptr, eW, ebi, ea, E, HID, HID);

  for (int i = 0; i < 2; ++i) {
    const float* linW = gat_lin_W  + (size_t)i * HID * HID;
    const float* edgW = gat_edge_W + (size_t)i * HID * HID;
    const float* as  = att_src  + (size_t)i * HID;
    const float* ad  = att_dst  + (size_t)i * HID;
    const float* ae  = att_edge + (size_t)i * HID;
    const float* gbi = gat_bias + (size_t)i * HID;
    const float* W1 = mlp_W1 + (size_t)i * HID * 3 * HID;
    const float* b1 = mlp_b1 + (size_t)i * HID;
    const float* W2 = mlp_W2 + (size_t)i * HID * HID;
    const float* b2 = mlp_b2 + (size_t)i * HID;
    const float* lg = ln_g + (size_t)i * HID;
    const float* lb = ln_b + (size_t)i * HID;

    fill_f32<<<gb((long)N * HID), 256, 0, stream>>>(loopA, 0.f, (long)N * HID);
    fill_f32<<<gb(N), 256, 0, stream>>>(deg, 0.f, N);
    fill_f32<<<gb((long)N * HID), 256, 0, stream>>>(gout, 0.f, (long)N * HID);
    fill_f32<<<gb((long)N * HEADS), 256, 0, stream>>>(den, 0.f, (long)N * HEADS);
    fill_u32<<<gb((long)N * HEADS), 256, 0, stream>>>(amaxU, ORD_NEG_INF, (long)N * HEADS);

    // self-loop attr = scatter-mean of incoming edge_attr
    scatter_loop_attr<<<gb((long)E * HID), 256, 0, stream>>>(ea, dst, loopA, deg, (long)E * HID);
    finalize_loop_attr<<<gb((long)N * HID), 256, 0, stream>>>(loopA, deg, (long)N * HID);

    // hh = h @ lin_W^T  (no bias)
    gemm_kernel<false, false, false><<<ngrid, 256, 0, stream>>>(
        h, nullptr, nullptr, nullptr, nullptr, linW, zbias, hh, N, HID, HID);

    // fold att_edge into edge_W, then edge-attention logits
    wa_kernel<<<1, 256, 0, stream>>>(edgW, ae, Wa);
    att_edge_kernel<<<(E + 7) / 8, 256, 0, stream>>>(ea, E, Wa, alphaE);
    att_edge_kernel<<<(N + 7) / 8, 256, 0, stream>>>(loopA, N, Wa, alphaE + (size_t)E * HEADS);
    asd_kernel<<<gb((long)N * HEADS), 256, 0, stream>>>(hh, as, ad, asrc, adst, N);

    // segment softmax + aggregation
    alpha1_kernel<<<gb((long)(E + N) * HEADS), 256, 0, stream>>>(
        src, dst, asrc, adst, alphaE, alpha, amaxU, E, N);
    alpha2_kernel<<<gb((long)(E + N) * HEADS), 256, 0, stream>>>(
        dst, alpha, amaxU, den, E, N);
    message_kernel<<<gb((long)(E + N) * HID), 256, 0, stream>>>(
        src, dst, hh, alpha, den, gout, E, N);

    // h += relu(LN(gout + bias))
    node_update_kernel<<<(N + 7) / 8, 256, 0, stream>>>(h, gout, gbi, lg, lb, N);

    // edge MLP (gather-cat fused into WMMA GEMM), residual into ea
    gemm_kernel<true, true, false><<<egrid, 256, 0, stream>>>(
        nullptr, h, ea, src, dst, W1, b1, t, E, 3 * HID, HID);
    gemm_kernel<false, false, true><<<egrid, 256, 0, stream>>>(
        t, nullptr, nullptr, nullptr, nullptr, W2, b2, ea, E, HID, HID);
  }

  // classifier
  gemm_kernel<true, true, false><<<egrid, 256, 0, stream>>>(
      nullptr, h, ea, src, dst, cW1, cb1, t, E, 3 * HID, HID);
  gemm_kernel<false, true, false><<<egrid, 256, 0, stream>>>(
      t, nullptr, nullptr, nullptr, nullptr, cW2, cb2, t2, E, HID, 32);
  final_dot_kernel<<<(E + 7) / 8, 256, 0, stream>>>(t2, cW3, cb3, out, E);
}
